// ResidualVectorQuantizer_59270548685249
// MI455X (gfx1250) — matrix operand order repaired and checked by reading, not compiled, not run on hardware
//
#include <hip/hip_runtime.h>
#include <hip/hip_bf16.h>

#define RVQ_N 65536
#define RVQ_D 256
#define RVQ_K 1024
#define RVQ_L 4
#define RVQ_BETA 0.25f

typedef __attribute__((ext_vector_type(16))) __bf16 v16bf;
typedef __attribute__((ext_vector_type(8)))  float  v8f;

// ---------------------------------------------------------------------------
// Zero the loss accumulator (graph-replay safe; d_ws is not re-poisoned).
// ---------------------------------------------------------------------------
__global__ void rvq_zero_kernel(float* __restrict__ lossAcc) {
  *lossAcc = 0.0f;
}

// ---------------------------------------------------------------------------
// Codebook prep: bf16 copy + per-entry squared norms. One wave per codebook row.
// ---------------------------------------------------------------------------
__global__ void rvq_cbprep_kernel(const float* __restrict__ cb,  // [L*K, D]
                                  __bf16* __restrict__ cbB,
                                  float* __restrict__ cbNorm) {
  const int lane = threadIdx.x & 31;
  const int row  = blockIdx.x * 8 + (threadIdx.x >> 5);
  const float* src = cb  + (size_t)row * RVQ_D;
  __bf16*      dst = cbB + (size_t)row * RVQ_D;
  float ss = 0.0f;
#pragma unroll
  for (int t = 0; t < RVQ_D / 32; ++t) {
    int d = lane + t * 32;
    float v = src[d];
    ss += v * v;
    dst[d] = (__bf16)v;
  }
#pragma unroll
  for (int off = 16; off >= 1; off >>= 1) ss += __shfl_xor(ss, off, 32);
  if (lane == 0) cbNorm[row] = ss;
}

// ---------------------------------------------------------------------------
// Fused GEMM + argmin. Each wave owns 32 rows (two 16x16 C tiles that share
// every B fragment). argmin over k of (cbNorm[k] - 2 * r.cb_k); ||r||^2 is
// row-constant and cannot change the argmin. A fragments are built once per
// wave from the f32 residual (cvt to bf16 in registers -> no bf16 shadow
// array in HBM).
//
// Layouts used here (wave32; A/B K order is contiguous, not the ISA table order):
//   A 16x32 bf16 : lane l<16 row M=l holds K 0..15; l>=16 row M=l-16 K 16..31
//   B 32x16 bf16 : lane l<16 col N=l holds K 0..15; l>=16 col N=l-16 K 16..31
//   C 16x16 f32  : VGPR i: lanes 0-15 -> M=i, lanes 16-31 -> M=8+i; N=lane%16
// ---------------------------------------------------------------------------
__global__ void __launch_bounds__(256) rvq_argmin_kernel(
    const float*  __restrict__ rsrc,    // [N, D] f32 residual source (x at lv 0)
    const __bf16* __restrict__ cbB,     // [K, D] bf16 codebook (level slice)
    const float*  __restrict__ cbNorm,  // [K]
    int* __restrict__ codes) {          // [N]
  const int lane    = threadIdx.x & 31;
  const int wave    = threadIdx.x >> 5;
  const int l16     = lane & 15;
  const int half    = lane >> 4;
  const int rowBase = (blockIdx.x * 8 + wave) * 32;

  // Build A fragments for both row tiles (register-resident across all K).
  v16bf a0[8], a1[8];
  {
    const float* ap0 = rsrc + (size_t)(rowBase + l16) * RVQ_D;
    const float* ap1 = rsrc + (size_t)(rowBase + 16 + l16) * RVQ_D;
#pragma unroll
    for (int kk = 0; kk < 8; ++kk) {
      const int d0 = kk * 32 + half * 16;      // K = d0..d0+7
      const int d1 = kk * 32 + half * 16 + 8;  // K = d1..d1+7
      v8f lo0 = *(const v8f*)(ap0 + d0);
      v8f hi0 = *(const v8f*)(ap0 + d1);
      v8f lo1 = *(const v8f*)(ap1 + d0);
      v8f hi1 = *(const v8f*)(ap1 + d1);
#pragma unroll
      for (int i = 0; i < 8; ++i) {
        a0[kk][i] = (__bf16)lo0[i]; a0[kk][8 + i] = (__bf16)hi0[i];
        a1[kk][i] = (__bf16)lo1[i]; a1[kk][8 + i] = (__bf16)hi1[i];
      }
    }
  }

  float best0[8], best1[8];
  int   bidx0[8], bidx1[8];
#pragma unroll
  for (int i = 0; i < 8; ++i) {
    best0[i] = 3.402823e38f; best1[i] = 3.402823e38f;
    bidx0[i] = 0;            bidx1[i] = 0;
  }

  for (int ct = 0; ct < RVQ_K / 16; ++ct) {
    const int j = ct * 16 + l16;  // the code column this lane owns in C/D
    const __bf16* bp = cbB + (size_t)j * RVQ_D + half * 16;
    v8f acc0 = {}, acc1 = {};
#pragma unroll
    for (int kk = 0; kk < 8; ++kk) {
      v16bf b = *(const v16bf*)(bp + kk * 32);  // 16 contiguous bf16 of cb[j]
      acc0 = __builtin_amdgcn_wmma_f32_16x16x32_bf16(
          false, a0[kk], false, b, (short)0, acc0, false, false);
      acc1 = __builtin_amdgcn_wmma_f32_16x16x32_bf16(
          false, a1[kk], false, b, (short)0, acc1, false, false);
    }
    if (ct + 1 < RVQ_K / 16)  // uniform branch, EXEC untouched before WMMAs
      __builtin_prefetch(cbB + (size_t)(j + 16) * RVQ_D + half * 16, 0, 0);

    const float nb = cbNorm[j];
#pragma unroll
    for (int i = 0; i < 8; ++i) {
      float s0 = nb - 2.0f * acc0[i];
      float s1 = nb - 2.0f * acc1[i];
      if (s0 < best0[i]) { best0[i] = s0; bidx0[i] = j; }
      if (s1 < best1[i]) { best1[i] = s1; bidx1[i] = j; }
    }
  }

  // Argmin across the 16 lanes holding the same rows (lowest index on ties,
  // matching jnp.argmin). Butterfly leaves the result in every lane.
#pragma unroll
  for (int i = 0; i < 8; ++i) {
    float s0 = best0[i]; int i0 = bidx0[i];
    float s1 = best1[i]; int i1 = bidx1[i];
#pragma unroll
    for (int off = 8; off >= 1; off >>= 1) {
      float so = __shfl_xor(s0, off, 32); int io = __shfl_xor(i0, off, 32);
      if (so < s0 || (so == s0 && io < i0)) { s0 = so; i0 = io; }
      so = __shfl_xor(s1, off, 32); io = __shfl_xor(i1, off, 32);
      if (so < s1 || (so == s1 && io < i1)) { s1 = so; i1 = io; }
    }
    bidx0[i] = i0; bidx1[i] = i1;
  }
  if (l16 == 0) {
#pragma unroll
    for (int i = 0; i < 8; ++i) {
      codes[rowBase + half * 8 + i]      = bidx0[i];  // tile0: M = half*8 + i
      codes[rowBase + 16 + half * 8 + i] = bidx1[i];  // tile1
    }
  }
}

// ---------------------------------------------------------------------------
// Gather q = cb[idx], update residual (f32), accumulate qsum into d_out,
// write codes as floats, wave-reduced loss -> atomicAdd. One wave per row.
// Level 0 reads x and creates res; level 3 skips the dead res write.
// ---------------------------------------------------------------------------
__global__ void rvq_update_kernel(const float* __restrict__ cb,    // [K, D] level slice
                                  const int* __restrict__ codes,
                                  const float* __restrict__ rsrc,  // x (lv 0) or res
                                  float* __restrict__ res,
                                  float* __restrict__ qsum,        // d_out[0:N*D]
                                  float* __restrict__ codesOut,    // d_out tail [N*L]
                                  float* __restrict__ lossAcc,
                                  int level) {
  const int lane = threadIdx.x & 31;
  const int n    = blockIdx.x * 8 + (threadIdx.x >> 5);
  const int idx  = codes[n];
  const float* q = cb + (size_t)idx * RVQ_D;
  const size_t base = (size_t)n * RVQ_D;
  float ss = 0.0f;
#pragma unroll
  for (int t = 0; t < RVQ_D / 32; ++t) {
    int d = lane + t * 32;
    float rv = rsrc[base + d];
    float qv = q[d];
    float diff = rv - qv;
    ss += diff * diff;
    if (level != RVQ_L - 1) res[base + d] = diff;  // final residual unused
    float qs = qv;
    if (level != 0) qs += qsum[base + d];  // level 0 overwrites poisoned d_out
    qsum[base + d] = qs;
  }
#pragma unroll
  for (int off = 16; off >= 1; off >>= 1) ss += __shfl_xor(ss, off, 32);
  if (lane == 0) {
    codesOut[(size_t)n * RVQ_L + level] = (float)idx;
    atomicAdd(lossAcc, ss);
  }
}

__global__ void rvq_finalize_kernel(const float* __restrict__ lossAcc,
                                    float* __restrict__ lossOut) {
  // loss = sum over levels of (1+beta) * sum_sq / (N*D)
  *lossOut = lossAcc[0] * ((1.0f + RVQ_BETA) / ((float)RVQ_N * (float)RVQ_D));
}

// ---------------------------------------------------------------------------
extern "C" void kernel_launch(void* const* d_in, const int* in_sizes, int n_in,
                              void* d_out, int out_size, void* d_ws, size_t ws_size,
                              hipStream_t stream) {
  const float* x  = (const float*)d_in[0];  // [N, D]
  const float* cb = (const float*)d_in[1];  // [L, K, D]

  float* out       = (float*)d_out;
  float* quantized = out;                               // [N*D]
  float* codesOut  = out + (size_t)RVQ_N * RVQ_D;       // [N*L]
  float* lossOut   = codesOut + (size_t)RVQ_N * RVQ_L;  // [1]

  char* w = (char*)d_ws;
  float*  res    = (float*)w;  w += (size_t)RVQ_N * RVQ_D * 4;          // 64 MB
  __bf16* cbB    = (__bf16*)w; w += (size_t)RVQ_L * RVQ_K * RVQ_D * 2;  // 2 MB
  float*  cbNorm = (float*)w;  w += (size_t)RVQ_L * RVQ_K * 4;          // 16 KB
  int*    codes  = (int*)w;    w += (size_t)RVQ_N * 4;                  // 256 KB
  float*  lossAcc = (float*)w;

  rvq_zero_kernel<<<1, 1, 0, stream>>>(lossAcc);
  rvq_cbprep_kernel<<<(RVQ_L * RVQ_K) / 8, 256, 0, stream>>>(cb, cbB, cbNorm);

  for (int lv = 0; lv < RVQ_L; ++lv) {
    const float* rsrc = (lv == 0) ? x : res;
    rvq_argmin_kernel<<<RVQ_N / 256, 256, 0, stream>>>(
        rsrc, cbB + (size_t)lv * RVQ_K * RVQ_D, cbNorm + (size_t)lv * RVQ_K, codes);
    rvq_update_kernel<<<RVQ_N / 8, 256, 0, stream>>>(
        cb + (size_t)lv * RVQ_K * RVQ_D, codes, rsrc, res,
        quantized, codesOut, lossAcc, lv);
  }
  rvq_finalize_kernel<<<1, 1, 0, stream>>>(lossAcc, lossOut);
}
